// TransformerBlock_9431748182242
// MI455X (gfx1250) — matrix operand order repaired and checked by reading, not compiled, not run on hardware
//
#include <hip/hip_runtime.h>
#include <hip/hip_bf16.h>
#include <math.h>

typedef __attribute__((ext_vector_type(16))) _Float16 v16h;
typedef __attribute__((ext_vector_type(8)))  _Float16 v8h;
typedef __attribute__((ext_vector_type(8)))  float    v8f;
typedef __attribute__((ext_vector_type(4)))  float    v4f;

#define NB   4
#define DIMC 48
#define C3   144
#define HWT  65536LL   // 256*256
#define NTOT (NB * HWT)

__device__ __forceinline__ float gelu_exact(float x) {
  return 0.5f * x * (1.0f + erff(x * 0.70710678118654752f));
}

// Build a v16h WMMA fragment from two 16-byte (8-half) contiguous loads.
__device__ __forceinline__ v16h ld16_2(const _Float16* __restrict__ p0,
                                       const _Float16* __restrict__ p1) {
  v8h lo = *(const v8h*)p0;
  v8h hi = *(const v8h*)p1;
  return __builtin_shufflevector(lo, hi, 0, 1, 2, 3, 4, 5, 6, 7,
                                 8, 9, 10, 11, 12, 13, 14, 15);
}

// ---------------------------------------------------------------------------
// WMMA lane layout used here, contiguous k order for A and B (v_wmma_f32_16x16x32_f16, wave32), g = lane>>4, c = lane&15:
//  A row m = m0+c : halfs [0..15] = K kk+16g..+15 (one contiguous 32B run)
//  B col n = n0+c : halfs [0..15] = K kk+16g..+15 (one contiguous 32B run)
//  D: lane writes rows m0+8g+v (v=0..7) at col n -> 8 contiguous NHWC elements
// ---------------------------------------------------------------------------

// Generic pointwise GEMM: out[n, m] = sum_k W[m,k] * X[n,k]
// Each wave: one 16-wide n-tile x MT m-tiles (B fragment reused MT times,
// MT independent accumulator chains for matrix-pipe overlap).
template<int MPAD, int KPAD, int MT, bool OUT16, bool DOGELU>
__global__ void gemm_pw(const _Float16* __restrict__ W,
                        const _Float16* __restrict__ X,
                        float* __restrict__ Yf,
                        _Float16* __restrict__ Yh,
                        long long Ntot) {
  const int wave = threadIdx.x >> 5;
  const int lane = threadIdx.x & 31;
  const long long ntile = (long long)blockIdx.x * (blockDim.x >> 5) + wave;
  const long long n0 = ntile * 16;
  if (n0 >= Ntot) return;                 // wave-uniform
  const int g = lane >> 4;
  const int c = lane & 15;
  const long long n = n0 + c;
  const int mbase = blockIdx.y * MT * 16;

  const _Float16* __restrict__ Xrow = X + n * (long long)KPAD;

  v8f acc[MT];
#pragma unroll
  for (int t = 0; t < MT; ++t)
#pragma unroll
    for (int v = 0; v < 8; ++v) acc[t][v] = 0.f;

#pragma unroll
  for (int s = 0; s < KPAD / 32; ++s) {
    const int kk = s * 32;
    v16h b = ld16_2(Xrow + kk + 16 * g, Xrow + kk + 16 * g + 8);
#pragma unroll
    for (int t = 0; t < MT; ++t) {
      const _Float16* __restrict__ Wrow = W + (long long)(mbase + t * 16 + c) * KPAD;
      v16h a = ld16_2(Wrow + kk + 16 * g, Wrow + kk + 16 * g + 8);
      acc[t] = __builtin_amdgcn_wmma_f32_16x16x32_f16(false, a, false, b,
                                                      (short)0, acc[t], false, false);
    }
  }

#pragma unroll
  for (int t = 0; t < MT; ++t) {
    const long long idx0 = n * (long long)MPAD + mbase + t * 16 + 8 * g;
    if constexpr (OUT16) {
      v8h o;
#pragma unroll
      for (int v = 0; v < 8; ++v) {
        float q = acc[t][v];
        if constexpr (DOGELU) q = gelu_exact(q);
        o[v] = (_Float16)q;
      }
      *(v8h*)(Yh + idx0) = o;
    } else {
      v4f o0, o1;
#pragma unroll
      for (int v = 0; v < 4; ++v) {
        float q0 = acc[t][v], q1 = acc[t][v + 4];
        if constexpr (DOGELU) { q0 = gelu_exact(q0); q1 = gelu_exact(q1); }
        o0[v] = q0; o1[v] = q1;
      }
      *(v4f*)(Yf + idx0) = o0;
      *(v4f*)(Yf + idx0 + 4) = o1;
    }
  }
}

// ---------------------------------------------------------------------------
// Dense 3x3 conv (127->127 padded to 128) as 9-tap implicit GEMM + GELU.
// Interior tiles (wave-uniform test) skip all boundary cndmasks.
// ---------------------------------------------------------------------------
template<int MT>
__global__ void gemm_conv3(const _Float16* __restrict__ Wc,   // [9][128][128]
                           const _Float16* __restrict__ X,    // [N][128]
                           _Float16* __restrict__ Y,          // [N][128]
                           long long Ntot) {
  const int wave = threadIdx.x >> 5;
  const int lane = threadIdx.x & 31;
  const long long ntile = (long long)blockIdx.x * (blockDim.x >> 5) + wave;
  const long long n0 = ntile * 16;
  if (n0 >= Ntot) return;
  const int g = lane >> 4;
  const int c = lane & 15;
  const long long n = n0 + c;
  const int hw0 = (int)(n0 & (HWT - 1));
  const int py = hw0 >> 8;          // uniform: tile lies in one image row
  const int px0 = hw0 & 255;        // uniform: x of column 0 (multiple of 16)
  const bool interior = (py >= 1) && (py <= 254) && (px0 >= 16) && (px0 <= 224);
  const int mbase = blockIdx.y * MT * 16;

  v8f acc[MT];
#pragma unroll
  for (int t = 0; t < MT; ++t)
#pragma unroll
    for (int v = 0; v < 8; ++v) acc[t][v] = 0.f;

  if (interior) {
#pragma unroll
    for (int tap = 0; tap < 9; ++tap) {
      const int dy = tap / 3 - 1, dx = tap % 3 - 1;
      const _Float16* __restrict__ Xs = X + (n + dy * 256 + dx) * 128;
      const _Float16* __restrict__ Wt = Wc + (long long)tap * 128 * 128;
#pragma unroll
      for (int s = 0; s < 4; ++s) {
        const int kk = s * 32;
        v16h b = ld16_2(Xs + kk + 16 * g, Xs + kk + 16 * g + 8);
#pragma unroll
        for (int t = 0; t < MT; ++t) {
          const _Float16* __restrict__ Wrow = Wt + (long long)(mbase + t * 16 + c) * 128;
          v16h a = ld16_2(Wrow + kk + 16 * g, Wrow + kk + 16 * g + 8);
          acc[t] = __builtin_amdgcn_wmma_f32_16x16x32_f16(false, a, false, b,
                                                          (short)0, acc[t], false, false);
        }
      }
    }
  } else {
    const int bidx = (int)(n >> 16);
    const int px = px0 + c;
    const v8h zz = {(_Float16)0.f, (_Float16)0.f, (_Float16)0.f, (_Float16)0.f,
                    (_Float16)0.f, (_Float16)0.f, (_Float16)0.f, (_Float16)0.f};
#pragma unroll
    for (int tap = 0; tap < 9; ++tap) {
      const int dy = tap / 3 - 1, dx = tap % 3 - 1;
      const int yy = py + dy, xx = px + dx;
      const bool valid = ((unsigned)yy < 256u) && ((unsigned)xx < 256u);
      const long long ns = valid ? (((long long)bidx << 16) + (yy * 256 + xx)) : n;
      const _Float16* __restrict__ Xs = X + ns * 128;
      const _Float16* __restrict__ Wt = Wc + (long long)tap * 128 * 128;
#pragma unroll
      for (int s = 0; s < 4; ++s) {
        const int kk = s * 32;
        v8h lo = *(const v8h*)(Xs + kk + 16 * g);
        v8h hi = *(const v8h*)(Xs + kk + 16 * g + 8);
        lo = valid ? lo : zz;     // v_cndmask, no exec branching
        hi = valid ? hi : zz;
        v16h b = __builtin_shufflevector(lo, hi, 0, 1, 2, 3, 4, 5, 6, 7,
                                         8, 9, 10, 11, 12, 13, 14, 15);
#pragma unroll
        for (int t = 0; t < MT; ++t) {
          const _Float16* __restrict__ Wrow = Wt + (long long)(mbase + t * 16 + c) * 128;
          v16h a = ld16_2(Wrow + kk + 16 * g, Wrow + kk + 16 * g + 8);
          acc[t] = __builtin_amdgcn_wmma_f32_16x16x32_f16(false, a, false, b,
                                                          (short)0, acc[t], false, false);
        }
      }
    }
  }

#pragma unroll
  for (int t = 0; t < MT; ++t) {
    v8h o;
#pragma unroll
    for (int v = 0; v < 8; ++v) o[v] = (_Float16)gelu_exact(acc[t][v]);
    *(v8h*)(Y + n * 128 + mbase + t * 16 + 8 * g) = o;
  }
}

// ---------------------------------------------------------------------------
// Weight packing (f32 -> zero-padded f16)
// ---------------------------------------------------------------------------
__global__ void k_pack(const float* __restrict__ src, _Float16* __restrict__ dst,
                       int M, int K, int Mpad, int Kstore) {
  int i = blockIdx.x * blockDim.x + threadIdx.x;
  if (i >= Mpad * Kstore) return;
  int m = i / Kstore, k = i % Kstore;
  float v = (m < M && k < K) ? src[m * K + k] : 0.f;
  dst[i] = (_Float16)v;
}

// ffn_dw_w (127,127,3,3) -> [9][128][128]
__global__ void k_pack_conv(const float* __restrict__ src, _Float16* __restrict__ dst) {
  int i = blockIdx.x * blockDim.x + threadIdx.x;
  if (i >= 9 * 128 * 128) return;
  int tap = i / (128 * 128);
  int r = i % (128 * 128);
  int m = r / 128, k = r % 128;
  float v = (m < 127 && k < 127) ? src[(m * 127 + k) * 9 + tap] : 0.f;
  dst[i] = (_Float16)v;
}

// ---------------------------------------------------------------------------
// LayerNorm over channels: x NCHW f32 -> f16 NHWC [N][64] (48 valid + 16 zero)
// ---------------------------------------------------------------------------
__global__ void k_ln1(const float* __restrict__ x, const float* __restrict__ w,
                      const float* __restrict__ bb, _Float16* __restrict__ out,
                      long long Ntot) {
  long long n = (long long)blockIdx.x * blockDim.x + threadIdx.x;
  if (n >= Ntot) return;
  int bidx = (int)(n >> 16);
  int hw = (int)(n & (HWT - 1));
  const float* xp = x + (long long)bidx * DIMC * HWT + hw;
  float v[DIMC];
  float mu = 0.f;
#pragma unroll
  for (int cc = 0; cc < DIMC; ++cc) { v[cc] = xp[(long long)cc * HWT]; mu += v[cc]; }
  mu *= (1.f / DIMC);
  float var = 0.f;
#pragma unroll
  for (int cc = 0; cc < DIMC; ++cc) { float d = v[cc] - mu; var += d * d; }
  var *= (1.f / DIMC);
  float rs = rsqrtf(var + 1e-5f);
#pragma unroll
  for (int cc = 0; cc < DIMC; ++cc)
    out[n * 64 + cc] = (_Float16)(((v[cc] - mu) * rs) * w[cc] + bb[cc]);
#pragma unroll
  for (int cc = DIMC; cc < 64; ++cc) out[n * 64 + cc] = (_Float16)0.f;
}

// ---------------------------------------------------------------------------
// Fused depthwise 3/5/7 convs; regroup into q/k/v concat buffers
// NHWC f16 stride 160 (144 valid + 16 zero pad).
// ---------------------------------------------------------------------------
__global__ void k_dwconv(const _Float16* __restrict__ qkv,   // [N][144]
                         const float* __restrict__ w3, const float* __restrict__ w5,
                         const float* __restrict__ w7,
                         _Float16* __restrict__ qc, _Float16* __restrict__ kc,
                         _Float16* __restrict__ vc, long long Ntot) {
  long long t = (long long)blockIdx.x * blockDim.x + threadIdx.x;
  if (t >= Ntot * DIMC) return;
  int c = (int)(t % DIMC);
  long long n = t / DIMC;
  int bidx = (int)(n >> 16);
  int hw = (int)(n & (HWT - 1));
  int py = hw >> 8, px = hw & 255;
  float acc[3][3];  // [part p][size s]
#pragma unroll
  for (int p = 0; p < 3; ++p)
#pragma unroll
    for (int s = 0; s < 3; ++s) acc[p][s] = 0.f;

  for (int ty = -3; ty <= 3; ++ty) {
    for (int tx = -3; tx <= 3; ++tx) {
      int yy = py + ty, xx = px + tx;
      float in_p[3] = {0.f, 0.f, 0.f};
      if ((unsigned)yy < 256u && (unsigned)xx < 256u) {
        long long ns = ((long long)bidx << 16) + (long long)(yy * 256 + xx);
        const _Float16* src = qkv + ns * C3;
        in_p[0] = (float)src[c];
        in_p[1] = (float)src[DIMC + c];
        in_p[2] = (float)src[2 * DIMC + c];
      }
      if (ty >= -1 && ty <= 1 && tx >= -1 && tx <= 1) {
        int wi = (ty + 1) * 3 + (tx + 1);
#pragma unroll
        for (int p = 0; p < 3; ++p) acc[p][0] += w3[(p * DIMC + c) * 9 + wi] * in_p[p];
      }
      if (ty >= -2 && ty <= 2 && tx >= -2 && tx <= 2) {
        int wi = (ty + 2) * 5 + (tx + 2);
#pragma unroll
        for (int p = 0; p < 3; ++p) acc[p][1] += w5[(p * DIMC + c) * 25 + wi] * in_p[p];
      }
      {
        int wi = (ty + 3) * 7 + (tx + 3);
#pragma unroll
        for (int p = 0; p < 3; ++p) acc[p][2] += w7[(p * DIMC + c) * 49 + wi] * in_p[p];
      }
    }
  }
#pragma unroll
  for (int s = 0; s < 3; ++s) {
    qc[n * 160 + s * DIMC + c] = (_Float16)acc[0][s];
    kc[n * 160 + s * DIMC + c] = (_Float16)acc[1][s];
    vc[n * 160 + s * DIMC + c] = (_Float16)acc[2][s];
  }
  if (c < 16) {  // zero the K-padding channels 144..159
    qc[n * 160 + 144 + c] = (_Float16)0.f;
    kc[n * 160 + 144 + c] = (_Float16)0.f;
    vc[n * 160 + 144 + c] = (_Float16)0.f;
  }
}

// ---------------------------------------------------------------------------
// Row L2 norms over spatial dim. grid.x = 2*NB*48 (q rows then k rows)
// ---------------------------------------------------------------------------
__global__ void k_norm(const float* __restrict__ qf, const float* __restrict__ kf,
                       float* __restrict__ norms) {
  int id = blockIdx.x;
  int which = id / (NB * DIMC);
  int rem = id % (NB * DIMC);
  int b = rem / DIMC, cc = rem % DIMC;
  const float* src = which ? kf : qf;
  float s = 0.f;
  for (int hw = threadIdx.x; hw < (int)HWT; hw += blockDim.x) {
    float v = src[((long long)b * HWT + hw) * DIMC + cc];
    s += v * v;
  }
  __shared__ float red[256];
  red[threadIdx.x] = s;
  __syncthreads();
  for (int st = 128; st > 0; st >>= 1) {
    if ((int)threadIdx.x < st) red[threadIdx.x] += red[threadIdx.x + st];
    __syncthreads();
  }
  if (threadIdx.x == 0) norms[id] = fmaxf(sqrtf(red[0]), 1e-12f);
}

// ---------------------------------------------------------------------------
// Per (b,head) 6x6 gram + temp scale + softmax. grid.x = NB*8
// ---------------------------------------------------------------------------
__global__ void k_gram(const float* __restrict__ qf, const float* __restrict__ kf,
                       const float* __restrict__ norms, const float* __restrict__ temp,
                       float* __restrict__ attn) {
  int bh = blockIdx.x;
  int b = bh >> 3, h = bh & 7;
  float acc[36];
#pragma unroll
  for (int j = 0; j < 36; ++j) acc[j] = 0.f;
  for (int hw = threadIdx.x; hw < (int)HWT; hw += blockDim.x) {
    long long base = ((long long)b * HWT + hw) * DIMC + h * 6;
    float q6[6], k6[6];
#pragma unroll
    for (int j = 0; j < 6; ++j) { q6[j] = qf[base + j]; k6[j] = kf[base + j]; }
#pragma unroll
    for (int cc = 0; cc < 6; ++cc)
#pragma unroll
      for (int d = 0; d < 6; ++d) acc[cc * 6 + d] += q6[cc] * k6[d];
  }
  __shared__ float red[256 * 36];
#pragma unroll
  for (int j = 0; j < 36; ++j) red[threadIdx.x * 36 + j] = acc[j];
  __syncthreads();
  for (int st = 128; st > 0; st >>= 1) {
    if ((int)threadIdx.x < st)
      for (int j = 0; j < 36; ++j)
        red[threadIdx.x * 36 + j] += red[(threadIdx.x + st) * 36 + j];
    __syncthreads();
  }
  if (threadIdx.x == 0) {
    float nq[6], nk[6];
#pragma unroll
    for (int j = 0; j < 6; ++j) {
      nq[j] = norms[b * DIMC + h * 6 + j];
      nk[j] = norms[NB * DIMC + b * DIMC + h * 6 + j];
    }
    float t = temp[h];
    for (int cc = 0; cc < 6; ++cc) {
      float sv[6], mx = -1e30f;
      for (int d = 0; d < 6; ++d) {
        sv[d] = red[cc * 6 + d] / (nq[cc] * nk[d]) * t;
        mx = fmaxf(mx, sv[d]);
      }
      float sum = 0.f, e[6];
      for (int d = 0; d < 6; ++d) { e[d] = expf(sv[d] - mx); sum += e[d]; }
      for (int d = 0; d < 6; ++d) attn[bh * 36 + cc * 6 + d] = e[d] / sum;
    }
  }
}

// out[n, h*6+c] = sum_d attn[b,h,c,d] * v[n, h*6+d]  -> f16 NHWC stride 64
__global__ void k_attnv(const float* __restrict__ vf, const float* __restrict__ attn,
                        _Float16* __restrict__ aout, long long Ntot) {
  long long n = (long long)blockIdx.x * blockDim.x + threadIdx.x;
  if (n >= Ntot) return;
  int b = (int)(n >> 16);
#pragma unroll
  for (int h = 0; h < 8; ++h) {
    const float* A = attn + (b * 8 + h) * 36;
    float v6[6];
#pragma unroll
    for (int d = 0; d < 6; ++d) v6[d] = vf[n * DIMC + h * 6 + d];
#pragma unroll
    for (int cc = 0; cc < 6; ++cc) {
      float o = 0.f;
#pragma unroll
      for (int d = 0; d < 6; ++d) o += A[cc * 6 + d] * v6[d];
      aout[n * 64 + h * 6 + cc] = (_Float16)o;
    }
  }
#pragma unroll
  for (int j = DIMC; j < 64; ++j) aout[n * 64 + j] = (_Float16)0.f;
}

// y1 = x + attn_out (residual), store f32 NHWC; LN2(y1) -> f16 NHWC stride 64
__global__ void k_addln2(const float* __restrict__ x, const float* __restrict__ pof,
                         const float* __restrict__ w, const float* __restrict__ bb,
                         float* __restrict__ y1f, _Float16* __restrict__ ln2h,
                         long long Ntot) {
  long long n = (long long)blockIdx.x * blockDim.x + threadIdx.x;
  if (n >= Ntot) return;
  int bidx = (int)(n >> 16);
  int hw = (int)(n & (HWT - 1));
  const float* xp = x + (long long)bidx * DIMC * HWT + hw;
  float v[DIMC];
  float mu = 0.f;
#pragma unroll
  for (int cc = 0; cc < DIMC; ++cc) {
    v[cc] = xp[(long long)cc * HWT] + pof[n * DIMC + cc];
    y1f[n * DIMC + cc] = v[cc];
    mu += v[cc];
  }
  mu *= (1.f / DIMC);
  float var = 0.f;
#pragma unroll
  for (int cc = 0; cc < DIMC; ++cc) { float d = v[cc] - mu; var += d * d; }
  var *= (1.f / DIMC);
  float rs = rsqrtf(var + 1e-5f);
#pragma unroll
  for (int cc = 0; cc < DIMC; ++cc)
    ln2h[n * 64 + cc] = (_Float16)(((v[cc] - mu) * rs) * w[cc] + bb[cc]);
#pragma unroll
  for (int cc = DIMC; cc < 64; ++cc) ln2h[n * 64 + cc] = (_Float16)0.f;
}

// d_out NCHW f32 = y1 + ffn_out (both f32 NHWC)
__global__ void k_final(const float* __restrict__ y1f, const float* __restrict__ ffnf,
                        float* __restrict__ out, long long Ntot) {
  long long n = (long long)blockIdx.x * blockDim.x + threadIdx.x;
  if (n >= Ntot) return;
  int bidx = (int)(n >> 16);
  int hw = (int)(n & (HWT - 1));
#pragma unroll
  for (int cc = 0; cc < DIMC; ++cc)
    out[(long long)bidx * DIMC * HWT + (long long)cc * HWT + hw] =
        y1f[n * DIMC + cc] + ffnf[n * DIMC + cc];
}

// ---------------------------------------------------------------------------
extern "C" void kernel_launch(void* const* d_in, const int* in_sizes, int n_in,
                              void* d_out, int out_size, void* d_ws, size_t ws_size,
                              hipStream_t stream) {
  (void)in_sizes; (void)n_in; (void)out_size; (void)ws_size;
  const float* x       = (const float*)d_in[0];
  const float* ln1_w   = (const float*)d_in[1];
  const float* ln1_b   = (const float*)d_in[2];
  const float* qkv_w   = (const float*)d_in[3];
  const float* dw3_w   = (const float*)d_in[4];
  const float* dw5_w   = (const float*)d_in[5];
  const float* dw7_w   = (const float*)d_in[6];
  const float* q_w     = (const float*)d_in[7];
  const float* k_w     = (const float*)d_in[8];
  const float* v_w     = (const float*)d_in[9];
  const float* po_w    = (const float*)d_in[10];
  const float* temp    = (const float*)d_in[11];
  const float* ln2_w   = (const float*)d_in[12];
  const float* ln2_b   = (const float*)d_in[13];
  const float* ffnin_w = (const float*)d_in[14];
  const float* ffndw_w = (const float*)d_in[15];
  const float* ffnout_w= (const float*)d_in[16];
  float* out = (float*)d_out;

  const long long Ntot = NTOT;
  char* ws = (char*)d_ws;
  size_t off = 0;
  auto take = [&](size_t bytes) -> char* {
    char* p = ws + off;
    off = (off + bytes + 255) & ~(size_t)255;
    return p;
  };

  _Float16* XLN  = (_Float16*)take(Ntot * 64 * 2);
  _Float16* QKV  = (_Float16*)take(Ntot * C3 * 2);
  _Float16* QC   = (_Float16*)take(Ntot * 160 * 2);
  _Float16* KC   = (_Float16*)take(Ntot * 160 * 2);
  _Float16* VC   = (_Float16*)take(Ntot * 160 * 2);
  float*    QF   = (float*)take(Ntot * DIMC * 4);
  float*    KF   = (float*)take(Ntot * DIMC * 4);
  float*    VF   = (float*)take(Ntot * DIMC * 4);
  float*    NRM  = (float*)take(2 * NB * DIMC * 4);
  float*    ATT  = (float*)take(NB * 8 * 36 * 4);
  _Float16* AOUT = (_Float16*)take(Ntot * 64 * 2);
  float*    POF  = (float*)take(Ntot * DIMC * 4);
  float*    Y1F  = (float*)take(Ntot * DIMC * 4);
  _Float16* LN2H = (_Float16*)take(Ntot * 64 * 2);
  _Float16* HIDH = (_Float16*)take(Ntot * 128 * 2);
  _Float16* GACT = (_Float16*)take(Ntot * 128 * 2);
  float*    FFNF = (float*)take(Ntot * DIMC * 4);
  _Float16* W_QKV = (_Float16*)take(144 * 64 * 2);
  _Float16* W_Q   = (_Float16*)take(48 * 160 * 2);
  _Float16* W_K   = (_Float16*)take(48 * 160 * 2);
  _Float16* W_V   = (_Float16*)take(48 * 160 * 2);
  _Float16* W_PO  = (_Float16*)take(48 * 64 * 2);
  _Float16* W_FI  = (_Float16*)take(128 * 64 * 2);
  _Float16* W_C   = (_Float16*)take(9 * 128 * 128 * 2);
  _Float16* W_FO  = (_Float16*)take(48 * 128 * 2);

  // ---- pack weights to f16 (zero padded) ----
  k_pack<<<(144 * 64 + 255) / 256, 256, 0, stream>>>(qkv_w, W_QKV, 144, 48, 144, 64);
  k_pack<<<(48 * 160 + 255) / 256, 256, 0, stream>>>(q_w, W_Q, 48, 144, 48, 160);
  k_pack<<<(48 * 160 + 255) / 256, 256, 0, stream>>>(k_w, W_K, 48, 144, 48, 160);
  k_pack<<<(48 * 160 + 255) / 256, 256, 0, stream>>>(v_w, W_V, 48, 144, 48, 160);
  k_pack<<<(48 * 64 + 255) / 256, 256, 0, stream>>>(po_w, W_PO, 48, 48, 48, 64);
  k_pack<<<(128 * 64 + 255) / 256, 256, 0, stream>>>(ffnin_w, W_FI, 127, 48, 128, 64);
  k_pack_conv<<<(9 * 128 * 128 + 255) / 256, 256, 0, stream>>>(ffndw_w, W_C);
  k_pack<<<(48 * 128 + 255) / 256, 256, 0, stream>>>(ffnout_w, W_FO, 48, 127, 48, 128);

  const int NPIX_BLOCKS = (int)(Ntot / 256);          // 1024
  const int NT = (int)(Ntot / 16);                     // 16384 n-tiles
  const int GX = NT / 8;                               // 8 waves/block

  // ---- attention branch ----
  k_ln1<<<NPIX_BLOCKS, 256, 0, stream>>>(x, ln1_w, ln1_b, XLN, Ntot);
  gemm_pw<144, 64, 3, true, false><<<dim3(GX, 3), 256, 0, stream>>>(
      W_QKV, XLN, nullptr, QKV, Ntot);
  k_dwconv<<<(int)((Ntot * DIMC) / 256), 256, 0, stream>>>(
      QKV, dw3_w, dw5_w, dw7_w, QC, KC, VC, Ntot);
  gemm_pw<48, 160, 3, false, false><<<dim3(GX, 1), 256, 0, stream>>>(
      W_Q, QC, QF, nullptr, Ntot);
  gemm_pw<48, 160, 3, false, false><<<dim3(GX, 1), 256, 0, stream>>>(
      W_K, KC, KF, nullptr, Ntot);
  gemm_pw<48, 160, 3, false, false><<<dim3(GX, 1), 256, 0, stream>>>(
      W_V, VC, VF, nullptr, Ntot);
  k_norm<<<2 * NB * DIMC, 256, 0, stream>>>(QF, KF, NRM);
  k_gram<<<NB * 8, 256, 0, stream>>>(QF, KF, NRM, temp, ATT);
  k_attnv<<<NPIX_BLOCKS, 256, 0, stream>>>(VF, ATT, AOUT, Ntot);
  gemm_pw<48, 64, 3, false, false><<<dim3(GX, 1), 256, 0, stream>>>(
      W_PO, AOUT, POF, nullptr, Ntot);

  // ---- FFN branch ----
  k_addln2<<<NPIX_BLOCKS, 256, 0, stream>>>(x, POF, ln2_w, ln2_b, Y1F, LN2H, Ntot);
  gemm_pw<128, 64, 4, true, false><<<dim3(GX, 2), 256, 0, stream>>>(
      W_FI, LN2H, nullptr, HIDH, Ntot);
  gemm_conv3<4><<<dim3(GX, 2), 256, 0, stream>>>(W_C, HIDH, GACT, Ntot);
  gemm_pw<48, 128, 3, false, false><<<dim3(GX, 1), 256, 0, stream>>>(
      W_FO, GACT, FFNF, nullptr, Ntot);

  k_final<<<NPIX_BLOCKS, 256, 0, stream>>>(Y1F, FFNF, out, Ntot);
}
